// GraphAggregator_21526376088205
// MI455X (gfx1250) — matrix operand order repaired and checked by reading, not compiled, not run on hardware
//
#include <hip/hip_runtime.h>
#include <hip/hip_bf16.h>

// ---------------------------------------------------------------------------
// Types for CDNA5 WMMA / TDM
// ---------------------------------------------------------------------------
typedef __attribute__((ext_vector_type(16))) __bf16 v16bf;
typedef __attribute__((ext_vector_type(8)))  __bf16 v8bf;
typedef __attribute__((ext_vector_type(8)))  float  v8f;
typedef __attribute__((ext_vector_type(4)))  unsigned int v4u;
typedef __attribute__((ext_vector_type(8)))  int    v8i;
typedef __attribute__((ext_vector_type(4)))  int    v4i;

#define NUM_GRAPHS 1024
#define IN_NODE    128
#define OUT_GRAPH  128
#define ROWS_PER_BLOCK 128
#define TILE_BYTES_F32 (ROWS_PER_BLOCK * IN_NODE * 4)   // 64 KB

#if defined(__has_builtin)
#  if __has_builtin(__builtin_amdgcn_tensor_load_to_lds)
#    define GA_HAS_TDM 1
#  endif
#  if __has_builtin(__builtin_amdgcn_s_wait_tensorcnt)
#    define GA_HAS_TWAIT 1
#  endif
#endif
#ifndef GA_HAS_TDM
#  define GA_HAS_TDM 0
#endif
#ifndef GA_HAS_TWAIT
#  define GA_HAS_TWAIT 0
#endif

static __device__ __forceinline__ unsigned short f32_to_bf16(float f) {
    unsigned int u = __float_as_uint(f);
    u += 0x7FFFu + ((u >> 16) & 1u);           // round to nearest even
    return (unsigned short)(u >> 16);
}
static __device__ __forceinline__ float bf16_to_f32(unsigned short h) {
    return __uint_as_float(((unsigned int)h) << 16);
}
static __device__ __forceinline__ __bf16 bfbits(float f) {
    unsigned short u = f32_to_bf16(f);
    return __builtin_bit_cast(__bf16, u);
}
static __device__ __forceinline__ v8f zero8() {
    v8f z;
#pragma unroll
    for (int i = 0; i < 8; ++i) z[i] = 0.0f;
    return z;
}
static __device__ __forceinline__ v8f wmma_bf16(v16bf a, v16bf b, v8f c) {
    return __builtin_amdgcn_wmma_f32_16x16x32_bf16(
        false, a, false, b, (short)0, c, false, false);
}

// A-fragment for 16x16x32 bf16 WMMA, built from an f32 LDS row with inline
// RNE conversion (each x element is converted exactly once).
// lane<16 : row M=lane,    elems 0..15 = K kb+0..15
// lane>=16: row M=lane-16, elems 0..15 = K kb+16..31
static __device__ __forceinline__ v16bf make_a_frag_f32(const float* row,
                                                        int kb, int khalf) {
    const float4 a0 = *(const float4*)(row + kb + 2 * khalf);
    const float4 a1 = *(const float4*)(row + kb + 2 * khalf + 4);
    const float4 b0 = *(const float4*)(row + kb + 2 * khalf + 8);
    const float4 b1 = *(const float4*)(row + kb + 2 * khalf + 12);
    v16bf a;
    a[0]  = bfbits(a0.x); a[1]  = bfbits(a0.y); a[2]  = bfbits(a0.z); a[3]  = bfbits(a0.w);
    a[4]  = bfbits(a1.x); a[5]  = bfbits(a1.y); a[6]  = bfbits(a1.z); a[7]  = bfbits(a1.w);
    a[8]  = bfbits(b0.x); a[9]  = bfbits(b0.y); a[10] = bfbits(b0.z); a[11] = bfbits(b0.w);
    a[12] = bfbits(b1.x); a[13] = bfbits(b1.y); a[14] = bfbits(b1.z); a[15] = bfbits(b1.w);
    return a;
}

// A-fragment from a bf16 LDS row (used by the final kernel).
static __device__ __forceinline__ v16bf load_a_frag_bf16(const unsigned short* row,
                                                         int kb, int khalf) {
    v8bf lo = *(const v8bf*)(row + kb + 2 * khalf);
    v8bf hi = *(const v8bf*)(row + kb + 2 * khalf + 8);
    v16bf a;
#pragma unroll
    for (int i = 0; i < 8; ++i) { a[i] = lo[i]; a[8 + i] = hi[i]; }
    return a;
}

#if GA_HAS_TDM
// Issue a TDM load of a [rows<=128, 128] f32 tile into LDS.
// D# layout per CDNA5 ISA.  tensor_dim1 = rows
// remaining so TDM's OOB-read-returns-zero zero-pads the tail tile.
static __device__ __forceinline__ void tdm_load_tile(const float* gsrc,
                                                     unsigned lds_byte_off,
                                                     unsigned rowsRemaining) {
    unsigned long long ga = (unsigned long long)(uintptr_t)gsrc;
    v4u g0;
    g0[0] = 1u;                                        // count=1, user mode
    g0[1] = lds_byte_off;                              // lds_addr
    g0[2] = (unsigned)(ga & 0xFFFFFFFFu);              // global_addr[31:0]
    g0[3] = (unsigned)((ga >> 32) & 0x01FFFFFFu)       // global_addr[56:32]
            | (2u << 30);                              // type = 2 ("image")
    const unsigned td1 = rowsRemaining;
    v8i g1;
    g1[0] = (int)(2u << 16);                           // data_size = 4 bytes
    g1[1] = (int)((unsigned)IN_NODE << 16);            // tensor_dim0[15:0]
    g1[2] = (int)((td1 & 0xFFFFu) << 16);              // td0 hi=0 | td1 lo
    g1[3] = (int)(((td1 >> 16) & 0xFFFFu)              // td1 hi
            | ((unsigned)IN_NODE << 16));              // tile_dim0 = 128
    g1[4] = (int)ROWS_PER_BLOCK;                       // tile_dim1=128, tile_dim2=0
    g1[5] = (int)IN_NODE;                              // tensor_dim0_stride lo
    g1[6] = 0;                                         // stride0 hi | stride1 lo
    g1[7] = 0;
    v4i z4 = {0, 0, 0, 0};
#if __clang_major__ >= 23
    v8i z8 = {0, 0, 0, 0, 0, 0, 0, 0};
    __builtin_amdgcn_tensor_load_to_lds(g0, g1, z4, z4, z8, 0);
#else
    __builtin_amdgcn_tensor_load_to_lds(g0, g1, z4, z4, 0);
#endif
}
#endif

static __device__ __forceinline__ void ga_wait_tensor(int leaveOutstanding) {
#if GA_HAS_TDM
#if GA_HAS_TWAIT
    if (leaveOutstanding) __builtin_amdgcn_s_wait_tensorcnt(1);
    else                  __builtin_amdgcn_s_wait_tensorcnt(0);
#else
    if (leaveOutstanding) asm volatile("s_wait_tensorcnt 0x1" ::: "memory");
    else                  asm volatile("s_wait_tensorcnt 0x0" ::: "memory");
#endif
#else
    (void)leaveOutstanding;
#endif
}

// ---------------------------------------------------------------------------
// Kernel 1: convert weights f32->bf16, zero accumulators
// ---------------------------------------------------------------------------
__global__ void ga_prep_kernel(const float* __restrict__ Wl,
                               const float* __restrict__ Wg,
                               const float* __restrict__ Wf,
                               unsigned short* __restrict__ WlB,
                               unsigned short* __restrict__ WgB,
                               unsigned short* __restrict__ WfB,
                               float* __restrict__ sums,
                               float* __restrict__ counts) {
    int i = blockIdx.x * blockDim.x + threadIdx.x;
    int stride = gridDim.x * blockDim.x;
    for (int j = i; j < IN_NODE * OUT_GRAPH; j += stride) {
        WlB[j] = f32_to_bf16(Wl[j]);
        WgB[j] = f32_to_bf16(Wg[j]);
        WfB[j] = f32_to_bf16(Wf[j]);
    }
    for (int j = i; j < NUM_GRAPHS * OUT_GRAPH; j += stride) sums[j] = 0.0f;
    for (int j = i; j < NUM_GRAPHS; j += stride) counts[j] = 0.0f;
}

// ---------------------------------------------------------------------------
// Kernel 2: persistent, TDM double-buffered dual GEMM + softmax + gate +
//           sorted-segment run-length atomics.
//   256 threads = 8 waves; LDS = 2 x 64KB f32 tile buffers (dynamic).
// ---------------------------------------------------------------------------
__global__ void __launch_bounds__(256)
ga_main_kernel(const float* __restrict__ x,
               const int*   __restrict__ batch,
               const float* __restrict__ b_lin,
               const float* __restrict__ b_gate,
               const unsigned short* __restrict__ WlB,
               const unsigned short* __restrict__ WgB,
               float* __restrict__ sums,
               float* __restrict__ counts,
               int nNodes, int nTiles) {
    extern __shared__ __align__(16) unsigned char dynsmem[];
    __shared__ int sBatch[ROWS_PER_BLOCK];

    const int tid   = threadIdx.x;
    const int wave  = tid >> 5;
    const int lane  = tid & 31;
    const int lrow  = lane & 15;
    const int half  = lane >> 4;
    const int khalf = half ? 8 : 0;
    const int ksel  = half ? 16 : 0;
    const int rbase = wave * 16;

    int cur = blockIdx.x;
    if (cur >= nTiles) return;
    int bufIdx = 0;

#if GA_HAS_TDM
    // Prologue: DMA first tile into buffer 0 (wave 0 issues; duplicates under
    // EXEC quirks are benign, every wave waits its own TENSORcnt).
    if (wave == 0)
        tdm_load_tile(x + (long)cur * (ROWS_PER_BLOCK * IN_NODE),
                      (unsigned)(uintptr_t)(dynsmem),
                      (unsigned)(nNodes - cur * ROWS_PER_BLOCK));
#endif

    while (true) {
        const int nxt = cur + (int)gridDim.x;
        const int hasNext = (nxt < nTiles);
        float* bufCur = (float*)(dynsmem + bufIdx * TILE_BYTES_F32);

#if GA_HAS_TDM
        if (wave == 0 && hasNext)
            tdm_load_tile(x + (long)nxt * (ROWS_PER_BLOCK * IN_NODE),
                          (unsigned)(uintptr_t)(dynsmem + (bufIdx ^ 1) * TILE_BYTES_F32),
                          (unsigned)(nNodes - nxt * ROWS_PER_BLOCK));
        ga_wait_tensor(hasNext);      // <=1 outstanding => current tile done
#else
        // Fallback: synchronous cooperative staging of the current tile.
        {
#pragma unroll
            for (int it = 0; it < 16; ++it) {
                int i = tid + it * 256;
                int row = i >> 5, c4 = i & 31;
                long n = (long)cur * ROWS_PER_BLOCK + row;
                float4 v = make_float4(0.f, 0.f, 0.f, 0.f);
                if (n < nNodes) v = ((const float4*)x)[n * 32 + c4];
                ((float4*)bufCur)[i] = v;
            }
        }
#endif
        __syncthreads();              // current tile visible to all waves

        if (tid < ROWS_PER_BLOCK) {
            long n = (long)cur * ROWS_PER_BLOCK + tid;
            sBatch[tid] = (n < nNodes) ? batch[n] : -1;
        }

        // ---- dual GEMM: states (lin) + logits (gate), shared A fragment ----
        v8f accS[8], accG[8];
#pragma unroll
        for (int t = 0; t < 8; ++t) { accS[t] = zero8(); accG[t] = zero8(); }

        const float* aRow = bufCur + (rbase + lrow) * IN_NODE;
#pragma unroll
        for (int kk = 0; kk < 4; ++kk) {
            const int kb = kk * 32;
            v16bf a = make_a_frag_f32(aRow, kb, khalf);
#pragma unroll
            for (int t = 0; t < 8; ++t) {
                const int o = t * 16 + lrow;   // B: 16 contiguous K per lane
                v16bf bl = *(const v16bf*)(WlB + o * IN_NODE + kb + ksel);
                v16bf bg = *(const v16bf*)(WgB + o * IN_NODE + kb + ksel);
                accS[t] = wmma_bf16(a, bl, accS[t]);
                accG[t] = wmma_bf16(a, bg, accG[t]);
            }
        }

        // ---- bias ----
#pragma unroll
        for (int t = 0; t < 8; ++t) {
            const int f = t * 16 + lrow;
            const float bl = b_lin[f];
            const float bg = b_gate[f];
#pragma unroll
            for (int r = 0; r < 8; ++r) { accS[t][r] += bl; accG[t][r] += bg; }
        }

        // ---- row softmax over 128 features (xor masks stay in 16-lane half)
#pragma unroll
        for (int r = 0; r < 8; ++r) {
            float mx = -INFINITY;
#pragma unroll
            for (int t = 0; t < 8; ++t) mx = fmaxf(mx, accG[t][r]);
#pragma unroll
            for (int off = 1; off < 16; off <<= 1)
                mx = fmaxf(mx, __shfl_xor(mx, off, 32));
            float s = 0.0f;
            float ev[8];
#pragma unroll
            for (int t = 0; t < 8; ++t) { ev[t] = __expf(accG[t][r] - mx); s += ev[t]; }
#pragma unroll
            for (int off = 1; off < 16; off <<= 1)
                s += __shfl_xor(s, off, 32);
            const float inv = 1.0f / s;
#pragma unroll
            for (int t = 0; t < 8; ++t) accS[t][r] = accS[t][r] * ev[t] * inv;
        }

        __syncthreads();              // all waves done reading f32 tile
        // ---- spill gated tile as bf16 over the consumed buffer ----
        unsigned short* sGated = (unsigned short*)bufCur;
#pragma unroll
        for (int t = 0; t < 8; ++t) {
            const int col = t * 16 + lrow;
#pragma unroll
            for (int r = 0; r < 8; ++r) {
                const int row = rbase + r + 8 * half;
                sGated[row * OUT_GRAPH + col] = f32_to_bf16(accS[t][r]);
            }
        }
        __syncthreads();

        // ---- sorted batch -> run-length compressed segment atomics ----
        {
            const int f    = tid & 127;
            const int rbeg = (tid >> 7) * 64;
            float run = 0.0f, cnt = 0.0f;
            int curId = -1;
            for (int r = rbeg; r < rbeg + 64; ++r) {
                const int id = sBatch[r];
                if (id != curId) {
                    if (curId >= 0) {
                        atomicAdd(&sums[curId * OUT_GRAPH + f], run);
                        if (f == 0) atomicAdd(&counts[curId], cnt);
                    }
                    curId = id; run = 0.0f; cnt = 0.0f;
                }
                if (id >= 0) {
                    run += bf16_to_f32(sGated[r * OUT_GRAPH + f]);
                    cnt += 1.0f;
                }
            }
            if (curId >= 0) {
                atomicAdd(&sums[curId * OUT_GRAPH + f], run);
                if (f == 0) atomicAdd(&counts[curId], cnt);
            }
        }

        if (!hasNext) break;
        cur = nxt;
        bufIdx ^= 1;
        __syncthreads();              // scatter done before buffer is re-DMA'd
    }
}

// ---------------------------------------------------------------------------
// Kernel 3: mean = sums / clamp(counts,1); out = mean @ W_final^T + b_final
// ---------------------------------------------------------------------------
__global__ void __launch_bounds__(256)
ga_final_kernel(const float* __restrict__ sums,
                const float* __restrict__ counts,
                const unsigned short* __restrict__ WfB,
                const float* __restrict__ b_final,
                float* __restrict__ out) {
    __shared__ __align__(32) unsigned short sMean[16 * OUT_GRAPH];
    const int tid = threadIdx.x;
    const int g0  = blockIdx.x * 16;

#pragma unroll
    for (int j = 0; j < 8; ++j) {
        const int e  = tid * 8 + j;
        const int gl = e >> 7;
        const int f  = e & 127;
        const int g  = g0 + gl;
        const float c = counts[g];
        const float m = sums[g * OUT_GRAPH + f] / fmaxf(c, 1.0f);
        sMean[e] = f32_to_bf16(m);
    }
    __syncthreads();

    const int wave  = tid >> 5;
    const int lane  = tid & 31;
    const int lrow  = lane & 15;
    const int half  = lane >> 4;
    const int khalf = half ? 8 : 0;
    const int ksel  = half ? 16 : 0;

    v8f acc = zero8();
    const unsigned short* aRow = sMean + lrow * OUT_GRAPH;
#pragma unroll
    for (int kk = 0; kk < 4; ++kk) {
        const int kb = kk * 32;
        v16bf a = load_a_frag_bf16(aRow, kb, khalf);
        const int o = wave * 16 + lrow;
        v16bf b = *(const v16bf*)(WfB + o * OUT_GRAPH + kb + ksel);
        acc = wmma_bf16(a, b, acc);
    }

    const int o  = wave * 16 + lrow;
    const float bf = b_final[o];
#pragma unroll
    for (int r = 0; r < 8; ++r) {
        const int g = g0 + r + 8 * half;
        out[g * OUT_GRAPH + o] = acc[r] + bf;
    }
}

// ---------------------------------------------------------------------------
// Host launcher
// ---------------------------------------------------------------------------
extern "C" void kernel_launch(void* const* d_in, const int* in_sizes, int n_in,
                              void* d_out, int out_size, void* d_ws, size_t ws_size,
                              hipStream_t stream) {
    const float* x       = (const float*)d_in[0];
    const int*   batch   = (const int*)  d_in[1];
    const float* W_lin   = (const float*)d_in[2];
    const float* b_lin   = (const float*)d_in[3];
    const float* W_gate  = (const float*)d_in[4];
    const float* b_gate  = (const float*)d_in[5];
    const float* W_final = (const float*)d_in[6];
    const float* b_final = (const float*)d_in[7];
    float* out = (float*)d_out;

    const int nNodes = in_sizes[0] / IN_NODE;
    const int nTiles = (nNodes + ROWS_PER_BLOCK - 1) / ROWS_PER_BLOCK;

    char* ws = (char*)d_ws;
    float* sums   = (float*)ws;                               // 512 KB
    float* counts = (float*)(ws + 512 * 1024);                //   4 KB
    unsigned short* WlB = (unsigned short*)(ws + 516 * 1024); //  32 KB
    unsigned short* WgB = (unsigned short*)(ws + 548 * 1024); //  32 KB
    unsigned short* WfB = (unsigned short*)(ws + 580 * 1024); //  32 KB
    (void)n_in; (void)out_size; (void)ws_size;

    ga_prep_kernel<<<512, 256, 0, stream>>>(W_lin, W_gate, W_final,
                                            WlB, WgB, WfB, sums, counts);

    int nBlocks = nTiles < 1024 ? nTiles : 1024;   // persistent, ~2 blocks/WGP
    ga_main_kernel<<<nBlocks, 256, 2 * TILE_BYTES_F32, stream>>>(
        x, batch, b_lin, b_gate, WlB, WgB, sums, counts, nNodes, nTiles);

    ga_final_kernel<<<NUM_GRAPHS / 16, 256, 0, stream>>>(sums, counts, WfB,
                                                         b_final, out);
}
